// DynamicRewireGNN_65231963291901
// MI455X (gfx1250) — matrix operand order repaired and checked by reading, not compiled, not run on hardware
//
#include <hip/hip_runtime.h>
#include <hip/hip_bf16.h>
#include <stdint.h>

// Problem constants (from reference)
#define NN   50000
#define EE   800000
#define CC   400000
#define HH   64
#define LL   2
#define GG   128
#define NCLS 10

#define WPB  4   // waves per block
#define NTPW 4   // tiles per wave

typedef __attribute__((ext_vector_type(16))) __bf16          v16bf;
typedef __attribute__((ext_vector_type(16))) unsigned short  v16u;
typedef __attribute__((ext_vector_type(8)))  unsigned short  v8u;
typedef __attribute__((ext_vector_type(8)))  float           v8f;

union FragU { v16u u; v16bf b; v8u h8[2]; };

// f32 -> bf16 round-to-nearest-even
__device__ __forceinline__ unsigned short f2bf(float f) {
  unsigned int u = __float_as_uint(f);
  u += 0x7FFFu + ((u >> 16) & 1u);
  return (unsigned short)(u >> 16);
}

__device__ __forceinline__ v8f wmma_bf16(v16bf a, v16bf b, v8f c) {
  // (neg_a, A, neg_b, B, c_mod, C, reuse_a, reuse_b)
  return __builtin_amdgcn_wmma_f32_16x16x32_bf16(false, a, false, b, (short)0, c,
                                                 false, false);
}

// Load a pre-packed fragment (A or B): lane-major, 16 bf16 (32B) per lane.
__device__ __forceinline__ v16bf ld_frag(const unsigned short* p, int lane) {
  FragU f;
  f.u = *(const v16u*)(p + (size_t)lane * 16);
  return f.b;
}

// Build an A fragment from a row-major bf16 row (>=64 valid elements).
// A 16x32 layout: lane m=l&15; khalf=l>>4; K = kbase+khalf*8+[0..7],
// then kbase+16+khalf*8+[0..7]  -> two contiguous 16B loads.
__device__ __forceinline__ v16bf make_frag_bf(const unsigned short* rp, int kbase,
                                              int khalf) {
  FragU f;
  int b0 = kbase + khalf * 8;
  f.h8[0] = *(const v8u*)(rp + b0);
  f.h8[1] = *(const v8u*)(rp + b0 + 16);
  return f.b;
}

// Build an A fragment from an f32 row (with conversion).
__device__ __forceinline__ v16bf make_frag(const float* rp, int kbase, int khalf) {
  FragU f;
  int b0 = kbase + khalf * 8;
#pragma unroll
  for (int j = 0; j < 8; ++j) f.u[j]     = f2bf(rp[b0 + j]);
#pragma unroll
  for (int j = 0; j < 8; ++j) f.u[8 + j] = f2bf(rp[b0 + 16 + j]);
  return f.b;
}

// Same, but A row = p1[k] + p2[k] (fused h+agg).
__device__ __forceinline__ v16bf make_frag2(const float* p1, const float* p2,
                                            int kbase, int khalf) {
  FragU f;
  int b0 = kbase + khalf * 8;
#pragma unroll
  for (int j = 0; j < 8; ++j) f.u[j]     = f2bf(p1[b0 + j]      + p2[b0 + j]);
#pragma unroll
  for (int j = 0; j < 8; ++j) f.u[8 + j] = f2bf(p1[b0 + 16 + j] + p2[b0 + 16 + j]);
  return f.b;
}

// ---------------- packing kernels ----------------

// Pack KxNn f32 row-major weight into bf16 B panels (32x16 each).
// B layout: lane n=l&15; K = kb*32 + (l>>4)*8 + (j<8 ? j : j+8), j=0..15.
__global__ void k_pack_B(const float* __restrict__ W, int K, int Nn,
                         unsigned short* __restrict__ out) {
  int tid = blockIdx.x * blockDim.x + threadIdx.x;
  if (tid >= K * Nn) return;
  int p    = tid >> 9;
  int lane = (tid >> 4) & 31;
  int j    = tid & 15;
  int ntc  = Nn >> 4;
  int kb = p / ntc, nt = p % ntc;
  int k = kb * 32 + (lane >> 4) * 8 + (j < 8 ? j : j + 8);
  int n = nt * 16 + (lane & 15);
  out[tid] = f2bf(W[k * Nn + n]);
}

// Pack rowsx64 f32 (edge_attr) into bf16 A panels (16x32 each), 2 panels/tile.
__global__ void k_pack_A(const float* __restrict__ A, unsigned short* __restrict__ out,
                         long total) {
  long tid = (long)blockIdx.x * blockDim.x + threadIdx.x;
  if (tid >= total) return;
  long p   = tid >> 9;
  int lane = (int)((tid >> 4) & 31);
  int j    = (int)(tid & 15);
  long tile = p >> 1;
  int  kb   = (int)(p & 1);
  long row  = tile * 16 + (lane & 15);
  int  k    = kb * 32 + (lane >> 4) * 8 + (j < 8 ? j : j + 8);
  out[tid] = f2bf(A[row * 64 + k]);
}

__global__ void k_zero(float* p, long n) {
  long t = (long)blockIdx.x * blockDim.x + threadIdx.x;
  if (t < n) p[t] = 0.f;
}

// ---------------- GEMM kernels (1 wave = NTPW 16-row tiles) ----------------

// h = X @ W + b   (rows contiguous); also writes bf16 mirror of h.
__global__ void __launch_bounds__(128)
k_node_linear(const float* __restrict__ X,
              const unsigned short* __restrict__ Bp,
              const float* __restrict__ bias,
              float* __restrict__ Hout,
              unsigned short* __restrict__ Hbf, int ntiles) {
  int wave = threadIdx.x >> 5, lane = threadIdx.x & 31;
  int t0 = (blockIdx.x * (blockDim.x >> 5) + wave) * NTPW;
  if (t0 >= ntiles) return;
  int khalf = lane >> 4, ln = lane & 15;
  v16bf B[2][4];
#pragma unroll
  for (int kb = 0; kb < 2; ++kb)
#pragma unroll
    for (int nt = 0; nt < 4; ++nt) B[kb][nt] = ld_frag(Bp + (kb * 4 + nt) * 512, lane);
  for (int it = 0; it < NTPW; ++it) {
    int tile = t0 + it;
    if (tile >= ntiles) break;
    const float* rp = X + ((size_t)tile * 16 + ln) * 64;
    v16bf a0 = make_frag(rp, 0, khalf);
    v16bf a1 = make_frag(rp, 32, khalf);
#pragma unroll
    for (int nt = 0; nt < 4; ++nt) {
      int n = nt * 16 + ln;
      float bn = bias[n];
      v8f acc = {bn, bn, bn, bn, bn, bn, bn, bn};
      acc = wmma_bf16(a0, B[0][nt], acc);
      acc = wmma_bf16(a1, B[1][nt], acc);
#pragma unroll
      for (int r = 0; r < 8; ++r) {
        size_t idx = ((size_t)tile * 16 + r + khalf * 8) * 64 + n;
        Hout[idx] = acc[r];
        Hbf[idx]  = f2bf(acc[r]);
      }
    }
  }
}

// e = relu(edge_attr@W + b + h[src]) * w[e];  atomicAdd into agg[dst]
__global__ void __launch_bounds__(128)
k_edge_msg(const unsigned short* __restrict__ eapack,
           const unsigned short* __restrict__ Bp,
           const float* __restrict__ bias,
           const float* __restrict__ wvec,
           const int* __restrict__ src, const int* __restrict__ dst,
           const float* __restrict__ h, float* __restrict__ agg,
           int ntiles) {
  int wave = threadIdx.x >> 5, lane = threadIdx.x & 31;
  int t0 = (blockIdx.x * (blockDim.x >> 5) + wave) * NTPW;
  if (t0 >= ntiles) return;
  int khalf = lane >> 4, ln = lane & 15;
  v16bf B[2][4];
#pragma unroll
  for (int kb = 0; kb < 2; ++kb)
#pragma unroll
    for (int nt = 0; nt < 4; ++nt) B[kb][nt] = ld_frag(Bp + (kb * 4 + nt) * 512, lane);
  for (int it = 0; it < NTPW; ++it) {
    int tile = t0 + it;
    if (tile >= ntiles) break;
    if (tile + 1 < ntiles)  // prefetch next tile's A panels (HBM stream)
      __builtin_prefetch(eapack + ((size_t)(tile + 1) * 2) * 512, 0, 1);
    v16bf a0 = ld_frag(eapack + ((size_t)tile * 2 + 0) * 512, lane);
    v16bf a1 = ld_frag(eapack + ((size_t)tile * 2 + 1) * 512, lane);
    int se[8], de[8]; float we[8];
#pragma unroll
    for (int r = 0; r < 8; ++r) {
      int e = tile * 16 + r + khalf * 8;
      se[r] = src[e]; de[r] = dst[e]; we[r] = wvec[e];
    }
#pragma unroll
    for (int nt = 0; nt < 4; ++nt) {
      int n = nt * 16 + ln;
      float bn = bias[n];
      v8f acc = {bn, bn, bn, bn, bn, bn, bn, bn};
      acc = wmma_bf16(a0, B[0][nt], acc);
      acc = wmma_bf16(a1, B[1][nt], acc);
#pragma unroll
      for (int r = 0; r < 8; ++r) {
        float v = acc[r] + h[(size_t)se[r] * 64 + n];
        v = fmaxf(v, 0.f) * we[r];
        atomicAdd(&agg[(size_t)de[r] * 64 + n], v);
      }
    }
  }
}

// mode 0: h = h + relu( relu((h+agg)@W1+b1) @ W2 + b2 )
// mode 1: h =           relu((h+agg)@W1+b1) @ W2 + b2
// Intermediate t staged in LDS as bf16; also maintains bf16 mirror of h.
// B fragments loaded per-tile (weights are L0/L2-resident) to avoid spills.
__global__ void __launch_bounds__(128)
k_mlp2(const float* __restrict__ Hbuf, const float* __restrict__ Agg,
       const unsigned short* __restrict__ B1, const float* __restrict__ b1,
       const unsigned short* __restrict__ B2, const float* __restrict__ b2,
       int ntiles, int mode, float* __restrict__ Hout,
       unsigned short* __restrict__ Hbf) {
  __shared__ unsigned short tl[WPB][16 * 72];
  int wave = threadIdx.x >> 5, lane = threadIdx.x & 31;
  int t0 = (blockIdx.x * (blockDim.x >> 5) + wave) * NTPW;
  unsigned short* tw = tl[wave];
  int khalf = lane >> 4, ln = lane & 15;
  for (int it = 0; it < NTPW; ++it) {
    int tile = t0 + it;
    bool act = tile < ntiles;
    if (act) {
      const float* p1 = Hbuf + ((size_t)tile * 16 + ln) * 64;
      const float* p2 = Agg  + ((size_t)tile * 16 + ln) * 64;
      v16bf a0 = make_frag2(p1, p2, 0, khalf);
      v16bf a1 = make_frag2(p1, p2, 32, khalf);
#pragma unroll
      for (int nt = 0; nt < 4; ++nt) {
        int n = nt * 16 + ln;
        float bn = b1[n];
        v8f acc = {bn, bn, bn, bn, bn, bn, bn, bn};
        acc = wmma_bf16(a0, ld_frag(B1 + (0 * 4 + nt) * 512, lane), acc);
        acc = wmma_bf16(a1, ld_frag(B1 + (1 * 4 + nt) * 512, lane), acc);
#pragma unroll
        for (int r = 0; r < 8; ++r)
          tw[(r + khalf * 8) * 72 + n] = f2bf(fmaxf(acc[r], 0.f));
      }
    }
    __syncthreads();
    if (act) {
      const unsigned short* tp = tw + ln * 72;
      v16bf a0 = make_frag_bf(tp, 0, khalf);
      v16bf a1 = make_frag_bf(tp, 32, khalf);
#pragma unroll
      for (int nt = 0; nt < 4; ++nt) {
        int n = nt * 16 + ln;
        float bn = b2[n];
        v8f acc = {bn, bn, bn, bn, bn, bn, bn, bn};
        acc = wmma_bf16(a0, ld_frag(B2 + (0 * 4 + nt) * 512, lane), acc);
        acc = wmma_bf16(a1, ld_frag(B2 + (1 * 4 + nt) * 512, lane), acc);
#pragma unroll
        for (int r = 0; r < 8; ++r) {
          size_t idx = ((size_t)tile * 16 + r + khalf * 8) * 64 + n;
          float v = (mode == 0) ? (Hbuf[idx] + fmaxf(acc[r], 0.f)) : acc[r];
          Hout[idx] = v;
          Hbf[idx]  = f2bf(v);
        }
      }
    }
    __syncthreads();
  }
}

// score = sigmoid( relu(concat(h[ia],h[ib])@W1 + b1) . W2 + b2 ) [* mulw]
// A fragments come straight from the bf16 mirror of h (no conversion VALU).
// B fragments loaded per-use (L0-resident) to keep VGPR pressure low.
__global__ void __launch_bounds__(128)
k_score(const int* __restrict__ ia, int sa,
        const int* __restrict__ ib, int sb,
        const unsigned short* __restrict__ Hbf,
        const unsigned short* __restrict__ B1, const float* __restrict__ b1,
        const float* __restrict__ W2, const float* __restrict__ b2p,
        const float* __restrict__ mulw, float* __restrict__ outw,
        int ntiles) {
  int wave = threadIdx.x >> 5, lane = threadIdx.x & 31;
  int t0 = (blockIdx.x * (blockDim.x >> 5) + wave) * NTPW;
  if (t0 >= ntiles) return;
  int khalf = lane >> 4, ln = lane & 15;
  float b2 = b2p[0];
  for (int it = 0; it < NTPW; ++it) {
    int tile = t0 + it;
    if (tile >= ntiles) break;
    long rowA = (long)tile * 16 + ln;
    const unsigned short* rpa = Hbf + (size_t)ia[rowA * sa] * 64;
    const unsigned short* rpb = Hbf + (size_t)ib[rowA * sb] * 64;
    v16bf a[4];
    a[0] = make_frag_bf(rpa, 0,  khalf);
    a[1] = make_frag_bf(rpa, 32, khalf);
    a[2] = make_frag_bf(rpb, 0,  khalf);
    a[3] = make_frag_bf(rpb, 32, khalf);
    float part[8] = {0, 0, 0, 0, 0, 0, 0, 0};
#pragma unroll
    for (int nt = 0; nt < 4; ++nt) {
      int n = nt * 16 + ln;
      float bn = b1[n];
      float w2 = W2[n];
      v8f acc = {bn, bn, bn, bn, bn, bn, bn, bn};
#pragma unroll
      for (int kb = 0; kb < 4; ++kb)
        acc = wmma_bf16(a[kb], ld_frag(B1 + (kb * 4 + nt) * 512, lane), acc);
#pragma unroll
      for (int r = 0; r < 8; ++r) part[r] += fmaxf(acc[r], 0.f) * w2;
    }
#pragma unroll
    for (int m = 1; m < 16; m <<= 1) {
#pragma unroll
      for (int r = 0; r < 8; ++r) part[r] += __shfl_xor(part[r], m, 32);
    }
    if (ln == 0) {
#pragma unroll
      for (int r = 0; r < 8; ++r) {
        long row = (long)tile * 16 + r + khalf * 8;
        float s  = part[r] + b2;
        float sg = 1.f / (1.f + __expf(-s));
        outw[row] = mulw ? sg * mulw[row] : sg;
      }
    }
  }
}

// m_add = relu(h[c0] + bias) * w_add; atomicAdd into agg[c1]
__global__ void k_add_msg(const int* __restrict__ cand, const float* __restrict__ bias,
                          const float* __restrict__ wadd, const float* __restrict__ h,
                          float* __restrict__ agg, long n64) {
  long t = (long)blockIdx.x * blockDim.x + threadIdx.x;
  if (t >= n64) return;
  long c = t >> 6; int j = (int)(t & 63);
  int a = cand[2 * c], b = cand[2 * c + 1];
  float v = fmaxf(h[(size_t)a * 64 + j] + bias[j], 0.f) * wadd[c];
  atomicAdd(&agg[(size_t)b * 64 + j], v);
}

__global__ void k_pool(const float* __restrict__ h, const int* __restrict__ batch,
                       float* __restrict__ pool, float* __restrict__ cnt, long n64) {
  long t = (long)blockIdx.x * blockDim.x + threadIdx.x;
  if (t >= n64) return;
  int i = (int)(t >> 6), j = (int)(t & 63);
  int g = batch[i];
  atomicAdd(&pool[(size_t)g * 64 + j], h[t]);
  if (j == 0) atomicAdd(&cnt[g], 1.0f);
}

__global__ void k_out(const float* __restrict__ pool, const float* __restrict__ cnt,
                      const float* __restrict__ W, const float* __restrict__ b,
                      float* __restrict__ out) {
  int t = blockIdx.x * blockDim.x + threadIdx.x;
  if (t >= GG * NCLS) return;
  int g = t / NCLS, c = t % NCLS;
  float inv = 1.0f / fmaxf(cnt[g], 1.0f);
  float s = 0.f;
#pragma unroll
  for (int k = 0; k < 64; ++k) s += pool[(size_t)g * 64 + k] * inv * W[k * NCLS + c];
  out[t] = s + b[c];
}

// ---------------- launch ----------------

extern "C" void kernel_launch(void* const* d_in, const int* in_sizes, int n_in,
                              void* d_out, int out_size, void* d_ws, size_t ws_size,
                              hipStream_t stream) {
  (void)in_sizes; (void)n_in; (void)out_size; (void)ws_size;

  const float* x            = (const float*)d_in[0];
  const float* edge_attr    = (const float*)d_in[1];
  const float* edge_weight  = (const float*)d_in[2];
  const int*   edge_index   = (const int*)d_in[3];
  const int*   cand         = (const int*)d_in[4];
  const int*   batch        = (const int*)d_in[5];
  const float* atom_W       = (const float*)d_in[6];
  const float* atom_b       = (const float*)d_in[7];
  const float* conv_edge_W  = (const float*)d_in[8];
  const float* conv_edge_b  = (const float*)d_in[9];
  const float* conv_W1      = (const float*)d_in[10];
  const float* conv_b1      = (const float*)d_in[11];
  const float* conv_W2      = (const float*)d_in[12];
  const float* conv_b2      = (const float*)d_in[13];
  const float* add_W1       = (const float*)d_in[14];
  const float* add_b1       = (const float*)d_in[15];
  const float* add_W2       = (const float*)d_in[16];
  const float* add_b2       = (const float*)d_in[17];
  const float* del_W1       = (const float*)d_in[18];
  const float* del_b1       = (const float*)d_in[19];
  const float* del_W2       = (const float*)d_in[20];
  const float* del_b2       = (const float*)d_in[21];
  const float* inter_edge_W = (const float*)d_in[22];
  const float* inter_edge_b = (const float*)d_in[23];
  const float* inter_W1     = (const float*)d_in[24];
  const float* inter_b1     = (const float*)d_in[25];
  const float* inter_W2     = (const float*)d_in[26];
  const float* inter_b2     = (const float*)d_in[27];
  const float* out_W        = (const float*)d_in[28];
  const float* out_b        = (const float*)d_in[29];

  const int* src = edge_index;
  const int* dst = edge_index + EE;

  uintptr_t base = (uintptr_t)d_ws;
  size_t off = 0;
  auto carve = [&](size_t bytes) -> void* {
    off = (off + 255) & ~(size_t)255;
    void* p = (void*)(base + off);
    off += bytes;
    return p;
  };
  float*          h      = (float*)carve((size_t)NN * 64 * 4);
  float*          agg    = (float*)carve((size_t)NN * 64 * 4);
  unsigned short* hbf    = (unsigned short*)carve((size_t)NN * 64 * 2);
  float*          wkeep  = (float*)carve((size_t)EE * 4);
  float*          wadd   = (float*)carve((size_t)CC * 4);
  float*          pool   = (float*)carve((size_t)GG * 64 * 4);
  float*          cnt    = (float*)carve((size_t)GG * 4);
  unsigned short* wpack  = (unsigned short*)carve((size_t)86016 * 2);
  unsigned short* eapack = (unsigned short*)carve((size_t)EE * 64 * 2);

  const int T = 256;
  auto gb = [](long n) { return (unsigned)((n + 255) / 256); };
  const int GT = WPB * 32;  // 128 threads, 4 waves
  auto tb = [&](int tiles) {
    return (unsigned)((tiles + WPB * NTPW - 1) / (WPB * NTPW));
  };

  // ---- one-time packing (weights + edge_attr) ----
  k_pack_B<<<gb(4096), T, 0, stream>>>(atom_W, 64, 64, wpack);
  for (int l = 0; l < LL; ++l) {
    unsigned short* lb = wpack + 4096 + l * 40960;
    k_pack_B<<<gb(4096), T, 0, stream>>>(conv_edge_W  + l * 4096, 64,  64, lb + 0);
    k_pack_B<<<gb(4096), T, 0, stream>>>(conv_W1      + l * 4096, 64,  64, lb + 4096);
    k_pack_B<<<gb(4096), T, 0, stream>>>(conv_W2      + l * 4096, 64,  64, lb + 8192);
    k_pack_B<<<gb(4096), T, 0, stream>>>(inter_edge_W + l * 4096, 64,  64, lb + 12288);
    k_pack_B<<<gb(4096), T, 0, stream>>>(inter_W1     + l * 4096, 64,  64, lb + 16384);
    k_pack_B<<<gb(4096), T, 0, stream>>>(inter_W2     + l * 4096, 64,  64, lb + 20480);
    k_pack_B<<<gb(8192), T, 0, stream>>>(add_W1       + l * 8192, 128, 64, lb + 24576);
    k_pack_B<<<gb(8192), T, 0, stream>>>(del_W1       + l * 8192, 128, 64, lb + 32768);
  }
  k_pack_A<<<gb((long)EE * 64), T, 0, stream>>>(edge_attr, eapack, (long)EE * 64);

  const int nt_n = NN / 16;  // 3125
  const int nt_e = EE / 16;  // 50000
  const int nt_c = CC / 16;  // 25000

  // h = x @ atom_W + atom_b  (+ bf16 mirror)
  k_node_linear<<<tb(nt_n), GT, 0, stream>>>(x, wpack, atom_b, h, hbf, nt_n);

  for (int l = 0; l < LL; ++l) {
    const unsigned short* lb = wpack + 4096 + l * 40960;

    // conv message + segment_sum
    k_zero<<<gb((long)NN * 64), T, 0, stream>>>(agg, (long)NN * 64);
    k_edge_msg<<<tb(nt_e), GT, 0, stream>>>(eapack, lb + 0, conv_edge_b + l * 64,
                                            edge_weight, src, dst, h, agg, nt_e);
    // h = h + relu(mlp2(h+agg))
    k_mlp2<<<tb(nt_n), GT, 0, stream>>>(h, agg, lb + 4096, conv_b1 + l * 64,
                                        lb + 8192, conv_b2 + l * 64, nt_n, 0, h, hbf);
    // add / del scores (A from bf16 mirror)
    k_score<<<tb(nt_c), GT, 0, stream>>>(cand, 2, cand + 1, 2, hbf,
                                         lb + 24576, add_b1 + l * 64, add_W2 + l * 64,
                                         add_b2 + l, nullptr, wadd, nt_c);
    k_score<<<tb(nt_e), GT, 0, stream>>>(src, 1, dst, 1, hbf,
                                         lb + 32768, del_b1 + l * 64, del_W2 + l * 64,
                                         del_b2 + l, edge_weight, wkeep, nt_e);
    // inter messages (keep + add) + segment_sums
    k_zero<<<gb((long)NN * 64), T, 0, stream>>>(agg, (long)NN * 64);
    k_edge_msg<<<tb(nt_e), GT, 0, stream>>>(eapack, lb + 12288, inter_edge_b + l * 64,
                                            wkeep, src, dst, h, agg, nt_e);
    k_add_msg<<<gb((long)CC * 64), T, 0, stream>>>(cand, inter_edge_b + l * 64, wadd,
                                                   h, agg, (long)CC * 64);
    // h = mlp2(h+agg2)
    k_mlp2<<<tb(nt_n), GT, 0, stream>>>(h, agg, lb + 16384, inter_b1 + l * 64,
                                        lb + 20480, inter_b2 + l * 64, nt_n, 1, h, hbf);
  }

  // mean-pool per graph + classifier head
  k_zero<<<gb((long)GG * 64), T, 0, stream>>>(pool, (long)GG * 64);
  k_zero<<<gb((long)GG), T, 0, stream>>>(cnt, (long)GG);
  k_pool<<<gb((long)NN * 64), T, 0, stream>>>(h, batch, pool, cnt, (long)NN * 64);
  k_out<<<gb((long)GG * NCLS), T, 0, stream>>>(pool, cnt, out_W, out_b, (float*)d_out);
}
